// GatedGCNConv_58969900974273
// MI455X (gfx1250) — matrix operand order repaired and checked by reading, not compiled, not run on hardware
//
#include <hip/hip_runtime.h>
#include <cstddef>
#include <cstdint>

// ---------------------------------------------------------------------------
// GatedGCN layer for MI455X (gfx1250, wave32, WMMA).
// - GEMMs: v_wmma_f32_16x16x32_f16, fp16 activations staged to LDS via
//   global_load_async_to_lds_b128 (ASYNCcnt path).
// - Elementwise/scatter kernels: wave-per-row float4 (b128) accesses,
//   block/wave-uniform index loads -> scalar loads.
// ---------------------------------------------------------------------------

constexpr int DIM = 128;

typedef __attribute__((ext_vector_type(16))) _Float16 v16h;
typedef __attribute__((ext_vector_type(8)))  _Float16 v8h;
typedef __attribute__((ext_vector_type(4)))  _Float16 v4h;
typedef __attribute__((ext_vector_type(8)))  float    v8f;
typedef __attribute__((ext_vector_type(4)))  float    v4f;

__device__ __forceinline__ void atomAddF(float* p, float v) {
  __hip_atomic_fetch_add(p, v, __ATOMIC_RELAXED, __HIP_MEMORY_SCOPE_AGENT);
}

// gfx1250 async global->LDS copy (16B per lane), tracked on ASYNCcnt.
__device__ __forceinline__ void async_g2l_b128(unsigned lds_off, const void* g) {
  asm volatile("global_load_async_to_lds_b128 %0, %1, off"
               :: "v"(lds_off), "v"((unsigned long long)(uintptr_t)g)
               : "memory");
}
__device__ __forceinline__ void wait_async0() {
  asm volatile("s_wait_asynccnt 0x0" ::: "memory");
}

// ---------------------------------------------------------------------------
// Utility kernels
// ---------------------------------------------------------------------------
__global__ __launch_bounds__(256) void zero_f32x4(v4f* __restrict__ p, size_t n4) {
  size_t i = (size_t)blockIdx.x * 256 + threadIdx.x;
  if (i < n4) p[i] = (v4f){0.f, 0.f, 0.f, 0.f};
}

// fp32 -> fp16 (vectorized), for activations.
__global__ __launch_bounds__(256) void cvt_f32_f16(const float* __restrict__ X,
                                                   _Float16* __restrict__ Y,
                                                   size_t n4) {
  size_t i = (size_t)blockIdx.x * 256 + threadIdx.x;
  if (i >= n4) return;
  v4f x = ((const v4f*)X)[i];
  v4h y;
#pragma unroll
  for (int t = 0; t < 4; ++t) y[t] = (_Float16)x[t];
  ((v4h*)Y)[i] = y;
}

// Weights [9][k][n] fp32 -> fp16 transposed [9][n][k] (B fragments = 32B loads).
__global__ __launch_bounds__(256) void cvt_w_t(const float* __restrict__ W,
                                               _Float16* __restrict__ Wt, int n) {
  int i = blockIdx.x * 256 + threadIdx.x;
  if (i >= n) return;
  int l = i >> 14;
  int k = (i >> 7) & 127;
  int c = i & 127;
  Wt[(size_t)l * 16384 + (size_t)c * 128 + k] = (_Float16)W[i];
}

// ---------------------------------------------------------------------------
// GEMM: Y[M x 128] = X16[M x 128] @ W + bias.  X16 fp16 row-major; Wt fp16 [n][k].
// 64 rows/block, 8 waves; wave w = column tile w; 4 row tiles; K=128 = 4 WMMAs.
// X tile staged to LDS with async b128 copies.
// ---------------------------------------------------------------------------
__global__ __launch_bounds__(256) void gemm128_bias(
    const _Float16* __restrict__ X, const _Float16* __restrict__ Wt,
    const float* __restrict__ bias, float* __restrict__ Y, int M) {
  __shared__ _Float16 xs[64 * DIM];  // 16 KB

  const int tid = threadIdx.x;
  const int wave = tid >> 5;
  const int lane = tid & 31;
  const int row0 = blockIdx.x * 64;
  const bool full = (row0 + 64 <= M);

  if (full) {
    // 16 KB tile: 256 threads x 64 B, async global->LDS.
    const char* gb = (const char*)(X + (size_t)row0 * DIM);
    unsigned lb = (unsigned)(uintptr_t)xs;
#pragma unroll
    for (int q = 0; q < 4; ++q) {
      unsigned o = (unsigned)tid * 64u + (unsigned)q * 16u;
      async_g2l_b128(lb + o, gb + o);
    }
    wait_async0();
  } else {
    for (int i = tid; i < 64 * DIM; i += 256) {
      int gr = row0 + (i >> 7);
      xs[i] = (gr < M) ? X[(size_t)gr * DIM + (i & 127)] : (_Float16)0.0f;
    }
  }
  __syncthreads();

  // B fragments (contiguous k order, same as A): lane -> col n; halves hold K 0..15 / 16..31.
  const int n = (lane & 15) + wave * 16;
  const int kb = (lane < 16) ? 0 : 16;
  v16h bfrag[4];
#pragma unroll
  for (int ks = 0; ks < 4; ++ks)
    bfrag[ks] = *(const v16h*)(Wt + (size_t)n * 128 + ks * 32 + kb);

  const float bval = bias[n];
  const int m_lo = lane & 15;
  const int ka = (lane < 16) ? 0 : 16;  // A halves: K 0..15 / 16..31

#pragma unroll
  for (int rt = 0; rt < 4; ++rt) {
    v8f acc;
#pragma unroll
    for (int r = 0; r < 8; ++r) acc[r] = bval;  // bias as C-init
#pragma unroll
    for (int ks = 0; ks < 4; ++ks) {
      const _Float16* xr = &xs[(rt * 16 + m_lo) * DIM + ks * 32 + ka];
      v8h lo = *(const v8h*)xr;
      v8h hi = *(const v8h*)(xr + 8);
      v16h a;
#pragma unroll
      for (int t = 0; t < 8; ++t) { a[t] = lo[t]; a[8 + t] = hi[t]; }
      acc = __builtin_amdgcn_wmma_f32_16x16x32_f16(
          false, a, false, bfrag[ks], (short)0, acc, false, false);
    }
    const int mb = rt * 16 + ((lane < 16) ? 0 : 8);
    if (full) {
      float* yb = Y + (size_t)(row0 + mb) * DIM + n;
#pragma unroll
      for (int r = 0; r < 8; ++r) yb[(size_t)r * DIM] = acc[r];
    } else {
#pragma unroll
      for (int r = 0; r < 8; ++r) {
        int grow = row0 + mb + r;
        if (grow < M) Y[(size_t)grow * DIM + n] = acc[r];
      }
    }
  }
}

// ---------------------------------------------------------------------------
// Wave-per-row float4 kernels (blockDim=128 = 4 waves; lane = 4 columns).
// ---------------------------------------------------------------------------

// e1 = Ah[src] + Ah[dst] + Be + Cu[a2g[src]] (in place); fused BN(1) stats.
__global__ __launch_bounds__(128) void edge_combine(
    const float* __restrict__ Ah, float* __restrict__ ePre,
    const float* __restrict__ Cu, const int* __restrict__ src,
    const int* __restrict__ dst, const int* __restrict__ a2g,
    float* __restrict__ stats, int rpb, int E) {
  const int lane = threadIdx.x & 31, wv = threadIdx.x >> 5;
  const int b0 = blockIdx.x * rpb, bend = min(b0 + rpb, E);
  const v4f* Ah4 = (const v4f*)Ah;
  const v4f* Cu4 = (const v4f*)Cu;
  v4f* e4 = (v4f*)ePre;
  v4f s1 = {0.f, 0.f, 0.f, 0.f}, s2 = s1;
  for (int b = b0 + wv; b < bend; b += 4) {
    int sa = src[b], da = dst[b], g = a2g[sa];
    v4f v = Ah4[(size_t)sa * 32 + lane] + Ah4[(size_t)da * 32 + lane] +
            e4[(size_t)b * 32 + lane] + Cu4[(size_t)g * 32 + lane];
    e4[(size_t)b * 32 + lane] = v;
    s1 += v;
    s2 += v * v;
  }
  const int j = lane * 4;
#pragma unroll
  for (int t = 0; t < 4; ++t) {
    atomAddF(&stats[j + t], s1[t]);
    atomAddF(&stats[DIM + j + t], s2[t]);
  }
}

// out = res + relu(bn(pre)); optional fp16 mirror of out.
__global__ __launch_bounds__(256) void bn_apply_residual(
    const float* __restrict__ res, const float* __restrict__ pre,
    float* __restrict__ out, _Float16* __restrict__ out16,
    const float* __restrict__ stats, const float* __restrict__ gamma,
    const float* __restrict__ beta, float invN, size_t n4) {
  size_t i = (size_t)blockIdx.x * 256 + threadIdx.x;
  if (i >= n4) return;
  const int j = (int)(i & 31) * 4;
  v4f p = ((const v4f*)pre)[i];
  v4f r = ((const v4f*)res)[i];
  v4f o;
  v4h o16;
#pragma unroll
  for (int t = 0; t < 4; ++t) {
    float m = stats[j + t] * invN;
    float var = fmaxf(stats[DIM + j + t] * invN - m * m, 0.0f);
    float sc = rsqrtf(var + 1e-5f) * gamma[j + t];
    float v = (p[t] - m) * sc + beta[j + t];
    float ov = r[t] + fmaxf(v, 0.0f);
    o[t] = ov;
    o16[t] = (_Float16)ov;
  }
  ((v4f*)out)[i] = o;
  if (out16) ((v4h*)out16)[i] = o16;
}

// num[dst] += sigmoid(e_out)*Eh[src]; den[dst] += sigmoid(e_out)
__global__ __launch_bounds__(128) void aggregate(
    const float* __restrict__ eOut, const float* __restrict__ Eh,
    const int* __restrict__ src, const int* __restrict__ dst,
    float* __restrict__ num, float* __restrict__ den, int rpb, int E) {
  const int lane = threadIdx.x & 31, wv = threadIdx.x >> 5;
  const int b0 = blockIdx.x * rpb, bend = min(b0 + rpb, E);
  const v4f* eo4 = (const v4f*)eOut;
  const v4f* Eh4 = (const v4f*)Eh;
  const int j = lane * 4;
  for (int b = b0 + wv; b < bend; b += 4) {
    int sa = src[b], da = dst[b];
    if (b + 4 < bend)  // global_prefetch_b8 of next gathered row
      __builtin_prefetch(&Eh4[(size_t)src[b + 4] * 32 + lane], 0, 0);
    v4f eo = eo4[(size_t)b * 32 + lane];
    v4f ev = Eh4[(size_t)sa * 32 + lane];
#pragma unroll
    for (int t = 0; t < 4; ++t) {
      float s = 1.0f / (1.0f + __expf(-eo[t]));
      atomAddF(&num[(size_t)da * DIM + j + t], s * ev[t]);
      atomAddF(&den[(size_t)da * DIM + j + t], s);
    }
  }
}

// hh = Dh + num/(den+eps) + Fu[a2g]; fused BN(0) stats + per-graph atom count.
__global__ __launch_bounds__(128) void node_combine(
    const float* __restrict__ Dh, const float* __restrict__ num,
    const float* __restrict__ den, const float* __restrict__ Fu,
    const int* __restrict__ a2g, float* __restrict__ hh,
    float* __restrict__ stats, float* __restrict__ cnt, int rpb, int NA) {
  const int lane = threadIdx.x & 31, wv = threadIdx.x >> 5;
  const int a0 = blockIdx.x * rpb, aend = min(a0 + rpb, NA);
  const v4f* Dh4 = (const v4f*)Dh;
  const v4f* nu4 = (const v4f*)num;
  const v4f* de4 = (const v4f*)den;
  const v4f* Fu4 = (const v4f*)Fu;
  v4f* hh4 = (v4f*)hh;
  v4f s1 = {0.f, 0.f, 0.f, 0.f}, s2 = s1;
  for (int a = a0 + wv; a < aend; a += 4) {
    int g = a2g[a];
    v4f nm = nu4[(size_t)a * 32 + lane];
    v4f dn = de4[(size_t)a * 32 + lane];
    v4f v = Dh4[(size_t)a * 32 + lane] + Fu4[(size_t)g * 32 + lane];
#pragma unroll
    for (int t = 0; t < 4; ++t) v[t] += nm[t] / (dn[t] + 1e-6f);
    hh4[(size_t)a * 32 + lane] = v;
    s1 += v;
    s2 += v * v;
    if (lane == 0) atomAddF(&cnt[g], 1.0f);
  }
  const int j = lane * 4;
#pragma unroll
  for (int t = 0; t < 4; ++t) {
    atomAddF(&stats[j + t], s1[t]);
    atomAddF(&stats[DIM + j + t], s2[t]);
  }
}

// He scatter onto graphs via a2g[dst] (random -> plain atomics).
__global__ __launch_bounds__(128) void scatter_edge_graph(
    const float* __restrict__ He, const int* __restrict__ dst,
    const int* __restrict__ a2g, float* __restrict__ HeG, int rpb, int E) {
  const int lane = threadIdx.x & 31, wv = threadIdx.x >> 5;
  const int b0 = blockIdx.x * rpb, bend = min(b0 + rpb, E);
  const v4f* He4 = (const v4f*)He;
  const int j = lane * 4;
  for (int b = b0 + wv; b < bend; b += 4) {
    int g = a2g[dst[b]];
    v4f v = He4[(size_t)b * 32 + lane];
#pragma unroll
    for (int t = 0; t < 4; ++t) atomAddF(&HeG[(size_t)g * DIM + j + t], v[t]);
  }
}

// Gh scatter via sorted a2g -> run-length local accumulation per wave.
__global__ __launch_bounds__(128) void scatter_atom_graph(
    const float* __restrict__ Gh, const int* __restrict__ a2g,
    float* __restrict__ GhSum, int rpb, int NA) {
  const int lane = threadIdx.x & 31, wv = threadIdx.x >> 5;
  const int a0 = blockIdx.x * rpb, aend = min(a0 + rpb, NA);
  const v4f* Gh4 = (const v4f*)Gh;
  const int j = lane * 4;
  v4f acc = {0.f, 0.f, 0.f, 0.f};
  int curg = -1;
  for (int a = a0 + wv; a < aend; a += 4) {
    int g = a2g[a];
    if (g != curg) {
      if (curg >= 0) {
#pragma unroll
        for (int t = 0; t < 4; ++t)
          atomAddF(&GhSum[(size_t)curg * DIM + j + t], acc[t]);
      }
      curg = g;
      acc = (v4f){0.f, 0.f, 0.f, 0.f};
    }
    acc += Gh4[(size_t)a * 32 + lane];
  }
  if (curg >= 0) {
#pragma unroll
    for (int t = 0; t < 4; ++t)
      atomAddF(&GhSum[(size_t)curg * DIM + j + t], acc[t]);
  }
}

// uu = GhSum/max(cnt,1) + HeG/E + Iu; fused BN(2) stats.
__global__ __launch_bounds__(128) void graph_combine(
    const float* __restrict__ GhSum, const float* __restrict__ HeG,
    const float* __restrict__ Iu, const float* __restrict__ cnt,
    float* __restrict__ uu, float* __restrict__ stats, float invE, int rpb,
    int NG) {
  const int lane = threadIdx.x & 31, wv = threadIdx.x >> 5;
  const int g0 = blockIdx.x * rpb, gend = min(g0 + rpb, NG);
  const v4f* Gs4 = (const v4f*)GhSum;
  const v4f* He4 = (const v4f*)HeG;
  const v4f* Iu4 = (const v4f*)Iu;
  v4f* uu4 = (v4f*)uu;
  v4f s1 = {0.f, 0.f, 0.f, 0.f}, s2 = s1;
  for (int g = g0 + wv; g < gend; g += 4) {
    float ic = 1.0f / fmaxf(cnt[g], 1.0f);
    v4f v = Iu4[(size_t)g * 32 + lane];
    v4f gs = Gs4[(size_t)g * 32 + lane];
    v4f he = He4[(size_t)g * 32 + lane];
#pragma unroll
    for (int t = 0; t < 4; ++t) v[t] += gs[t] * ic + he[t] * invE;
    uu4[(size_t)g * 32 + lane] = v;
    s1 += v;
    s2 += v * v;
  }
  const int j = lane * 4;
#pragma unroll
  for (int t = 0; t < 4; ++t) {
    atomAddF(&stats[j + t], s1[t]);
    atomAddF(&stats[DIM + j + t], s2[t]);
  }
}

// ---------------------------------------------------------------------------
// Host orchestration
// ---------------------------------------------------------------------------
extern "C" void kernel_launch(void* const* d_in, const int* in_sizes, int n_in,
                              void* d_out, int out_size, void* d_ws,
                              size_t ws_size, hipStream_t stream) {
  const float* h     = (const float*)d_in[0];
  const float* e     = (const float*)d_in[1];
  const float* u     = (const float*)d_in[2];
  const float* Ws    = (const float*)d_in[3];
  const float* bsv   = (const float*)d_in[4];
  const float* gamma = (const float*)d_in[5];
  const float* beta  = (const float*)d_in[6];
  const int*   src   = (const int*)d_in[7];
  const int*   dst   = (const int*)d_in[8];
  const int*   a2g   = (const int*)d_in[9];

  const int nA = in_sizes[0] / DIM;
  const int nE = in_sizes[1] / DIM;
  const int nG = in_sizes[2] / DIM;

  char* ws = (char*)d_ws;
  size_t off = 0;
  auto alloc = [&](size_t bytes) -> char* {
    char* p = ws + off;
    off += (bytes + 255) & ~(size_t)255;
    return p;
  };

  // zero region 1: stats + cnt + GhSum + HeG (contiguous)
  float* stats = (float*)alloc(3 * 2 * DIM * sizeof(float));
  float* cnt   = (float*)alloc((size_t)nG * sizeof(float));
  float* GhSum = (float*)alloc((size_t)nG * DIM * sizeof(float));
  float* HeG   = (float*)alloc((size_t)nG * DIM * sizeof(float));
  size_t zero1_f = ((char*)HeG + (size_t)nG * DIM * sizeof(float) -
                    (char*)stats) / sizeof(float);

  float* Cu = (float*)alloc((size_t)nG * DIM * sizeof(float));
  float* Fu = (float*)alloc((size_t)nG * DIM * sizeof(float));
  float* Iu = (float*)alloc((size_t)nG * DIM * sizeof(float));
  float* uu = (float*)alloc((size_t)nG * DIM * sizeof(float));
  _Float16* Wt  = (_Float16*)alloc((size_t)9 * DIM * DIM * sizeof(_Float16));
  _Float16* h16 = (_Float16*)alloc((size_t)nA * DIM * sizeof(_Float16));
  _Float16* e16 = (_Float16*)alloc((size_t)nE * DIM * sizeof(_Float16));
  _Float16* u16 = (_Float16*)alloc((size_t)nG * DIM * sizeof(_Float16));
  _Float16* ho16 = (_Float16*)alloc((size_t)nA * DIM * sizeof(_Float16));
  _Float16* eo16 = (_Float16*)alloc((size_t)nE * DIM * sizeof(_Float16));

  // zero region 2: num + den (contiguous)
  float* num = (float*)alloc((size_t)nA * DIM * sizeof(float));
  float* den = (float*)alloc((size_t)nA * DIM * sizeof(float));
  size_t zero2_f = ((char*)den + (size_t)nA * DIM * sizeof(float) -
                    (char*)num) / sizeof(float);

  float* Ah = (float*)alloc((size_t)nA * DIM * sizeof(float));  // later hh
  float* Eh = (float*)alloc((size_t)nA * DIM * sizeof(float));  // later Gh
  float* Dh = (float*)alloc((size_t)nA * DIM * sizeof(float));
  float* Bf = (float*)alloc((size_t)nE * DIM * sizeof(float));  // Be->e1->He
  float* hh = Ah;
  float* Gh = Eh;
  float* He = Bf;

  float* hOut = (float*)d_out;
  float* eOut = hOut + (size_t)nA * DIM;
  float* uOut = eOut + (size_t)nE * DIM;

  auto gemm = [&](const _Float16* X, int li, float* Y, int M) {
    gemm128_bias<<<dim3((M + 63) / 64), dim3(256), 0, stream>>>(
        X, Wt + (size_t)li * DIM * DIM, bsv + (size_t)li * DIM, Y, M);
  };
  auto cvt = [&](const float* X, _Float16* Y, size_t n4) {
    cvt_f32_f16<<<dim3((unsigned)((n4 + 255) / 256)), dim3(256), 0, stream>>>(
        X, Y, n4);
  };

  // ---- 0: zero accumulators ----
  zero_f32x4<<<dim3((unsigned)((zero1_f / 4 + 255) / 256)), dim3(256), 0,
               stream>>>((v4f*)stats, zero1_f / 4);
  zero_f32x4<<<dim3((unsigned)((zero2_f / 4 + 255) / 256)), dim3(256), 0,
               stream>>>((v4f*)num, zero2_f / 4);

  // ---- 1: precision conversions ----
  {
    int n = 9 * DIM * DIM;
    cvt_w_t<<<dim3((n + 255) / 256), dim3(256), 0, stream>>>(Ws, Wt, n);
  }
  cvt(h, h16, (size_t)nA * 32);
  cvt(e, e16, (size_t)nE * 32);
  cvt(u, u16, (size_t)nG * 32);

  // ---- 2: first-wave GEMMs (A,B,C,D,E,F) ----
  gemm(h16, 0, Ah, nA);
  gemm(e16, 1, Bf, nE);
  gemm(u16, 2, Cu, nG);
  gemm(h16, 3, Dh, nA);
  gemm(h16, 4, Eh, nA);
  gemm(u16, 5, Fu, nG);

  // ---- 3: edge update ----
  edge_combine<<<dim3((nE + 63) / 64), dim3(128), 0, stream>>>(
      Ah, Bf, Cu, src, dst, a2g, stats + 1 * 2 * DIM, 64, nE);
  bn_apply_residual<<<dim3((unsigned)(((size_t)nE * 32 + 255) / 256)),
                      dim3(256), 0, stream>>>(
      e, Bf, eOut, eo16, stats + 1 * 2 * DIM, gamma + DIM, beta + DIM,
      1.0f / (float)nE, (size_t)nE * 32);

  // ---- 4: gated aggregation ----
  aggregate<<<dim3((nE + 31) / 32), dim3(128), 0, stream>>>(
      eOut, Eh, src, dst, num, den, 32, nE);

  // ---- 5: node update ----
  node_combine<<<dim3((nA + 63) / 64), dim3(128), 0, stream>>>(
      Dh, num, den, Fu, a2g, hh, stats, cnt, 64, nA);
  bn_apply_residual<<<dim3((unsigned)(((size_t)nA * 32 + 255) / 256)),
                      dim3(256), 0, stream>>>(
      h, hh, hOut, ho16, stats, gamma, beta, 1.0f / (float)nA,
      (size_t)nA * 32);

  // ---- 6: second-wave GEMMs (G,H,I) ----
  gemm(ho16, 6, Gh, nA);
  gemm(eo16, 7, He, nE);
  gemm(u16, 8, Iu, nG);

  // ---- 7: graph update ----
  scatter_edge_graph<<<dim3((nE + 63) / 64), dim3(128), 0, stream>>>(
      He, dst, a2g, HeG, 64, nE);
  scatter_atom_graph<<<dim3((nA + 63) / 64), dim3(128), 0, stream>>>(
      Gh, a2g, GhSum, 64, nA);
  graph_combine<<<dim3((nG + 15) / 16), dim3(128), 0, stream>>>(
      GhSum, HeG, Iu, cnt, uu, stats + 2 * 2 * DIM, 1.0f / (float)nE, 16, nG);
  bn_apply_residual<<<dim3((unsigned)(((size_t)nG * 32 + 255) / 256)),
                      dim3(256), 0, stream>>>(
      u, uu, uOut, (_Float16*)nullptr, stats + 2 * 2 * DIM, gamma + 2 * DIM,
      beta + 2 * DIM, 1.0f / (float)nG, (size_t)nG * 32);
}
